// ParC_ConvNext_Block_65180423685138
// MI455X (gfx1250) — matrix operand order repaired and checked by reading, not compiled, not run on hardware
//
#include <hip/hip_runtime.h>
#include <hip/hip_bf16.h>
#include <math.h>

// ---------------------------------------------------------------------------
// ParC-ConvNeXt block for MI455X (gfx1250, wave32, WMMA).
//
// Roofline: MLP = 77 GFLOP dominates; min HBM traffic ~100MB (~4.5us).
// GEMMs on v_wmma_f32_16x16x32_bf16 (f32 accum). Fusion keeps the 201MB
// hidden tensor in LDS. M=64 tokens per WG => each B fragment (32B/lane)
// feeds 4 WMMAs (64 FLOP per L2 byte), ~1.2GB total L2 B-traffic.
// LDS strides padded (+8 elems) for conflict-free ds_load_b128.
// A/B operands software-pipelined (double-buffered) one K-step ahead so
// ds/global load latency is hidden behind the 4-WMMA group.
// ---------------------------------------------------------------------------

typedef __bf16 bf16;
typedef __attribute__((ext_vector_type(16))) __bf16 v16bf;
typedef __attribute__((ext_vector_type(8)))  __bf16 v8bf;
typedef __attribute__((ext_vector_type(8)))  float  v8f;

#define DIM   384
#define HID   1536            // 4*DIM
#define NTOK  32768           // B*H*W
#define KT1   12              // 384/32  K-tiles GEMM1
#define NT1   96              // 1536/16 N-tiles GEMM1
#define KT2   48              // 1536/32 K-tiles GEMM2
#define NT2   24              // 384/16  N-tiles GEMM2

#define MTOK  64              // tokens per workgroup (4 M-subtiles of 16)
#define SA_STRIDE 392         // 384 + 8 bf16 pad  (row shift = 4 dwords)
#define SH_STRIDE 1544        // 1536 + 8 bf16 pad
#define SA_BYTES  (MTOK * SA_STRIDE * 2)     // 50,176 B
#define SH_BYTES  (MTOK * SH_STRIDE * 2)     // 197,632 B
#define SMEM_BYTES (SA_BYTES + SH_BYTES)     // 247,808 B  (< 320KB WGP LDS)

union AB { v16bf v; v8bf h[2]; };

// load one 16x32 A fragment (two 16B LDS reads) for row block `row0+mrow`
static __device__ __forceinline__ AB load_a(const bf16* base, int stride,
                                            int row, int kb) {
    AB a;
    a.h[0] = *(const v8bf*)&base[(size_t)row * stride + kb];
    a.h[1] = *(const v8bf*)&base[(size_t)row * stride + kb + 16];
    return a;
}

// ---------------------------------------------------------------------------
// Pack w1 (384x1536) and w2 (1536x384) fp32 row-major into bf16 with the
// wave32 WMMA B-matrix layout for 16x16x32:
//   tile (kt,nt): lane L holds N = nt*16 + (L&15),
//                 elems e=0..15 -> K = kt*32 + (L>>4)*8 + (e&7) + (e>>3)*16
// ---------------------------------------------------------------------------
__global__ void pack_weights(const float* __restrict__ w1,
                             const float* __restrict__ w2,
                             bf16* __restrict__ w1p, bf16* __restrict__ w2p) {
    int tid = blockIdx.x * 256 + threadIdx.x;          // 1,179,648 total
    if (tid < DIM * HID) {
        int e = tid & 15, L = (tid >> 4) & 31, rest = tid >> 9;
        int kt = rest % KT1, nt = rest / KT1;
        int K = kt * 32 + ((L >> 4) & 1) * 8 + (e & 7) + (e >> 3) * 16;
        int N = nt * 16 + (L & 15);
        w1p[tid] = (bf16)w1[K * HID + N];
    } else {
        int t2 = tid - DIM * HID;
        int e = t2 & 15, L = (t2 >> 4) & 31, rest = t2 >> 9;
        int kt = rest % KT2, nt = rest / KT2;
        int K = kt * 32 + ((L >> 4) & 1) * 8 + (e & 7) + (e >> 3) * 16;
        int N = nt * 16 + (L & 15);
        w2p[t2] = (bf16)w2[K * DIM + N];
    }
}

// ---------------------------------------------------------------------------
// Fused ParC circular depthwise conv + channel LayerNorm.
// One WG per (b,h) row of 32 tokens; conv-H reads are lane-coalesced 128B
// lines (L2 resident), conv-W rows fully reused in the WG. LN via wave32
// shuffle reductions. Output: bf16 yln[token][384] (WMMA A operand source).
// ---------------------------------------------------------------------------
__global__ void __launch_bounds__(256) conv_ln(
        const float* __restrict__ x,
        const float* __restrict__ pe_h, const float* __restrict__ w_h,
        const float* __restrict__ b_h,
        const float* __restrict__ pe_w, const float* __restrict__ w_w,
        const float* __restrict__ b_w,
        const float* __restrict__ ln_w, const float* __restrict__ ln_b,
        bf16* __restrict__ yln) {
    __shared__ float yb[32 * DIM];                     // 48KB: [w][c]
    int bh = blockIdx.x;                               // b*32 + h
    int b  = bh >> 5, h = bh & 31;
    int w  = threadIdx.x & 31, cg = threadIdx.x >> 5;  // 8 channel groups

    for (int c = cg; c < DIM; c += 8) {
        float acc;
        if (c < 192) {                                  // circular conv along H
            acc = b_h[c];
            const float* xc = x + (((size_t)(b * DIM + c)) << 10) + w;
            #pragma unroll 8
            for (int i = 0; i < 32; ++i) {
                int hh = (h + i) & 31;
                acc += (xc[hh << 5] + pe_h[(c << 5) + hh]) * w_h[(c << 5) + i];
            }
        } else {                                        // circular conv along W
            int c2 = c - 192;
            acc = b_w[c2];
            const float* xr = x + (((size_t)(b * DIM + c)) << 10) + (h << 5);
            #pragma unroll 8
            for (int i = 0; i < 32; ++i) {
                int ww = (w + i) & 31;
                acc += (xr[ww] + pe_w[(c2 << 5) + ww]) * w_w[(c2 << 5) + i];
            }
        }
        yb[w * DIM + c] = acc;
    }
    __syncthreads();

    // LayerNorm: one wave per token (32 lanes x 12 channels each)
    int lane = threadIdx.x & 31, wave = threadIdx.x >> 5;
    for (int t0 = wave; t0 < 32; t0 += 8) {
        float v[12], s = 0.f;
        #pragma unroll
        for (int j = 0; j < 12; ++j) { v[j] = yb[t0 * DIM + j * 32 + lane]; s += v[j]; }
        #pragma unroll
        for (int m = 16; m >= 1; m >>= 1) s += __shfl_xor(s, m, 32);
        float mu = s * (1.0f / DIM);
        float sv = 0.f;
        #pragma unroll
        for (int j = 0; j < 12; ++j) { float d = v[j] - mu; sv += d * d; }
        #pragma unroll
        for (int m = 16; m >= 1; m >>= 1) sv += __shfl_xor(sv, m, 32);
        float rstd = rsqrtf(sv * (1.0f / DIM) + 1e-6f);
        size_t t = (size_t)bh * 32 + t0;
        #pragma unroll
        for (int j = 0; j < 12; ++j) {
            int c = j * 32 + lane;
            yln[t * DIM + c] = (bf16)(((v[j] - mu) * rstd) * ln_w[c] + ln_b[c]);
        }
    }
}

// ---------------------------------------------------------------------------
// Fused MLP: per WG tile of 64 tokens, 8 waves, dynamic LDS (242KB).
//  GEMM1 (K=384) -> exact GELU -> bf16 hidden in LDS -> GEMM2 (K=1536)
//  -> out = x + gamma*(acc + b2), NCHW fp32.
// ---------------------------------------------------------------------------
__global__ void __launch_bounds__(256, 1) mlp_fused(
        const bf16*  __restrict__ yln,
        const bf16*  __restrict__ w1p, const bf16* __restrict__ w2p,
        const float* __restrict__ b1,  const float* __restrict__ b2,
        const float* __restrict__ gamma,
        const float* __restrict__ xin, float* __restrict__ out) {
    extern __shared__ char smem[];
    bf16* sA = (bf16*)smem;                            // [64][SA_STRIDE]
    bf16* sH = (bf16*)(smem + SA_BYTES);               // [64][SH_STRIDE]
    const int tokBase = blockIdx.x * MTOK;

    // cooperative 16B-vector load: 64 rows x (2 x 24) uint4 (row = 384 bf16)
    for (int i = threadIdx.x; i < MTOK * 24; i += 256) {
        int row = i / 24, seg = i % 24;
        ((uint4*)(sA + (size_t)row * SA_STRIDE))[seg] =
            ((const uint4*)(yln + (size_t)(tokBase + row) * DIM))[seg];
        ((uint4*)(sA + (size_t)row * SA_STRIDE))[seg + 24] =
            ((const uint4*)(yln + (size_t)(tokBase + row) * DIM))[seg + 24];
    }
    __syncthreads();

    const int lane = threadIdx.x & 31, wave = threadIdx.x >> 5;
    const int hi = lane >> 4, mrow = lane & 15;

    // ---- GEMM1 + GELU: 12 N-tiles per wave x 4 M-subtiles x 12 K-steps ----
    for (int nt = wave; nt < NT1; nt += 8) {
        v8f acc0 = {}, acc1 = {}, acc2 = {}, acc3 = {};
        const bf16* bp = w1p + ((size_t)nt * KT1 * 32 + lane) * 16;
        // pipeline prologue: operands for kt = 0
        v16bf bm = *(const v16bf*)bp;
        int kb0 = hi * 8;
        AB a0 = load_a(sA, SA_STRIDE, mrow +  0, kb0);
        AB a1 = load_a(sA, SA_STRIDE, mrow + 16, kb0);
        AB a2 = load_a(sA, SA_STRIDE, mrow + 32, kb0);
        AB a3 = load_a(sA, SA_STRIDE, mrow + 48, kb0);
        #pragma unroll
        for (int kt = 0; kt < KT1 - 1; ++kt) {
            // issue next K-step's loads before consuming current operands
            v16bf bn = *(const v16bf*)(bp + (size_t)(kt + 1) * 32 * 16);
            int kb = (kt + 1) * 32 + hi * 8;
            AB n0 = load_a(sA, SA_STRIDE, mrow +  0, kb);
            AB n1 = load_a(sA, SA_STRIDE, mrow + 16, kb);
            AB n2 = load_a(sA, SA_STRIDE, mrow + 32, kb);
            AB n3 = load_a(sA, SA_STRIDE, mrow + 48, kb);
            acc0 = __builtin_amdgcn_wmma_f32_16x16x32_bf16(false, a0.v, false, bm, (short)0, acc0, false, false);
            acc1 = __builtin_amdgcn_wmma_f32_16x16x32_bf16(false, a1.v, false, bm, (short)0, acc1, false, false);
            acc2 = __builtin_amdgcn_wmma_f32_16x16x32_bf16(false, a2.v, false, bm, (short)0, acc2, false, false);
            acc3 = __builtin_amdgcn_wmma_f32_16x16x32_bf16(false, a3.v, false, bm, (short)0, acc3, false, false);
            a0 = n0; a1 = n1; a2 = n2; a3 = n3; bm = bn;
        }
        acc0 = __builtin_amdgcn_wmma_f32_16x16x32_bf16(false, a0.v, false, bm, (short)0, acc0, false, false);
        acc1 = __builtin_amdgcn_wmma_f32_16x16x32_bf16(false, a1.v, false, bm, (short)0, acc1, false, false);
        acc2 = __builtin_amdgcn_wmma_f32_16x16x32_bf16(false, a2.v, false, bm, (short)0, acc2, false, false);
        acc3 = __builtin_amdgcn_wmma_f32_16x16x32_bf16(false, a3.v, false, bm, (short)0, acc3, false, false);

        int col = nt * 16 + mrow;                      // C layout: N = lane&15
        float bias = b1[col];
        v8f* accs[4] = { &acc0, &acc1, &acc2, &acc3 };
        #pragma unroll
        for (int mt = 0; mt < 4; ++mt) {
            #pragma unroll
            for (int r = 0; r < 8; ++r) {              // M = r + hi*8
                float vv = (*accs[mt])[r] + bias;
                vv = 0.5f * vv * (1.0f + erff(vv * 0.70710678118654752f));
                sH[(size_t)(mt * 16 + r + hi * 8) * SH_STRIDE + col] = (bf16)vv;
            }
        }
    }
    __syncthreads();

    // ---- GEMM2 + residual: 3 N-tiles per wave x 4 M-subtiles x 48 K-steps --
    for (int nt = wave; nt < NT2; nt += 8) {
        v8f acc0 = {}, acc1 = {}, acc2 = {}, acc3 = {};
        const bf16* bp = w2p + ((size_t)nt * KT2 * 32 + lane) * 16;
        v16bf bm = *(const v16bf*)bp;
        int kb0 = hi * 8;
        AB a0 = load_a(sH, SH_STRIDE, mrow +  0, kb0);
        AB a1 = load_a(sH, SH_STRIDE, mrow + 16, kb0);
        AB a2 = load_a(sH, SH_STRIDE, mrow + 32, kb0);
        AB a3 = load_a(sH, SH_STRIDE, mrow + 48, kb0);
        #pragma unroll 4
        for (int kt = 0; kt < KT2 - 1; ++kt) {
            v16bf bn = *(const v16bf*)(bp + (size_t)(kt + 1) * 32 * 16);
            int kb = (kt + 1) * 32 + hi * 8;
            AB n0 = load_a(sH, SH_STRIDE, mrow +  0, kb);
            AB n1 = load_a(sH, SH_STRIDE, mrow + 16, kb);
            AB n2 = load_a(sH, SH_STRIDE, mrow + 32, kb);
            AB n3 = load_a(sH, SH_STRIDE, mrow + 48, kb);
            acc0 = __builtin_amdgcn_wmma_f32_16x16x32_bf16(false, a0.v, false, bm, (short)0, acc0, false, false);
            acc1 = __builtin_amdgcn_wmma_f32_16x16x32_bf16(false, a1.v, false, bm, (short)0, acc1, false, false);
            acc2 = __builtin_amdgcn_wmma_f32_16x16x32_bf16(false, a2.v, false, bm, (short)0, acc2, false, false);
            acc3 = __builtin_amdgcn_wmma_f32_16x16x32_bf16(false, a3.v, false, bm, (short)0, acc3, false, false);
            a0 = n0; a1 = n1; a2 = n2; a3 = n3; bm = bn;
        }
        acc0 = __builtin_amdgcn_wmma_f32_16x16x32_bf16(false, a0.v, false, bm, (short)0, acc0, false, false);
        acc1 = __builtin_amdgcn_wmma_f32_16x16x32_bf16(false, a1.v, false, bm, (short)0, acc1, false, false);
        acc2 = __builtin_amdgcn_wmma_f32_16x16x32_bf16(false, a2.v, false, bm, (short)0, acc2, false, false);
        acc3 = __builtin_amdgcn_wmma_f32_16x16x32_bf16(false, a3.v, false, bm, (short)0, acc3, false, false);

        int c = nt * 16 + mrow;
        float b2c = b2[c], gc = gamma[c];
        v8f* accs[4] = { &acc0, &acc1, &acc2, &acc3 };
        #pragma unroll
        for (int mt = 0; mt < 4; ++mt) {
            #pragma unroll
            for (int r = 0; r < 8; ++r) {
                int t = tokBase + mt * 16 + r + hi * 8;       // token index
                int bb = t >> 10, hw = t & 1023;
                size_t idx = (((size_t)(bb * DIM + c)) << 10) + hw;   // NCHW
                out[idx] = xin[idx] + gc * ((*accs[mt])[r] + b2c);
            }
        }
    }
}

// ---------------------------------------------------------------------------
extern "C" void kernel_launch(void* const* d_in, const int* in_sizes, int n_in,
                              void* d_out, int out_size, void* d_ws, size_t ws_size,
                              hipStream_t stream) {
    const float* x    = (const float*)d_in[0];
    const float* pe_h = (const float*)d_in[1];
    const float* w_h  = (const float*)d_in[2];
    const float* b_h  = (const float*)d_in[3];
    const float* pe_w = (const float*)d_in[4];
    const float* w_w  = (const float*)d_in[5];
    const float* b_w  = (const float*)d_in[6];
    const float* ln_w = (const float*)d_in[7];
    const float* ln_b = (const float*)d_in[8];
    const float* w1   = (const float*)d_in[9];
    const float* b1   = (const float*)d_in[10];
    const float* w2   = (const float*)d_in[11];
    const float* b2   = (const float*)d_in[12];
    const float* gamma= (const float*)d_in[13];
    float* out = (float*)d_out;

    // workspace: yln bf16 (25.2MB) | w1p bf16 (1.18MB) | w2p bf16 (1.18MB)
    char* ws = (char*)d_ws;
    bf16* yln = (bf16*)ws;
    bf16* w1p = (bf16*)(ws + (size_t)NTOK * DIM * 2);
    bf16* w2p = (bf16*)(ws + (size_t)NTOK * DIM * 2 + (size_t)DIM * HID * 2);

    pack_weights<<<(2 * DIM * HID) / 256, 256, 0, stream>>>(w1, w2, w1p, w2p);
    conv_ln<<<NTOK / 32, 256, 0, stream>>>(x, pe_h, w_h, b_h, pe_w, w_w, b_w,
                                           ln_w, ln_b, yln);
    mlp_fused<<<NTOK / MTOK, 256, SMEM_BYTES, stream>>>(yln, w1p, w2p, b1, b2,
                                                        gamma, x, out);
}
